// LinearMoELayer_45655502356775
// MI455X (gfx1250) — compile-verified, matrix operand order checked by reading, not run on hardware
//
#include <hip/hip_runtime.h>

// Problem constants (from reference)
#define BB 4
#define SS 2048
#define DD 2048
#define OO 2048
#define EE 8
#define TT (BB * SS)          // 8192 tokens
#define TOPK 2
#define NASSIGN (TT * TOPK)   // 16384 assignments
#define BAL_W 1e-2f
#define CV_EPS 1e-10f

#define D2 (DD / 2)           // row length in packed-bf16 uints
#define NSTEP (D2 / 32)       // K-steps of 64 bf16 (32 uints)

typedef __attribute__((ext_vector_type(16))) __bf16 v16bf;
typedef __attribute__((ext_vector_type(8)))  float  v8f;

// CDNA5-specific data movers (guarded; fall back to sync staging if absent)
#if defined(__has_builtin)
# if __has_builtin(__builtin_amdgcn_global_load_async_to_lds_b128) && \
     __has_builtin(__builtin_amdgcn_s_wait_asynccnt)
#  define HAVE_ASYNC 1
# endif
# if __has_builtin(__builtin_amdgcn_tensor_load_to_lds) && \
     __has_builtin(__builtin_amdgcn_s_wait_tensorcnt)
#  define HAVE_TDM 1
# endif
#endif
#ifndef HAVE_ASYNC
# define HAVE_ASYNC 0
#endif
#ifndef HAVE_TDM
# define HAVE_TDM 0
#endif

#define AS1 __attribute__((address_space(1)))
#define AS3 __attribute__((address_space(3)))

// gcc-style vectors to match the builtin prototypes exactly
typedef int      v4i_g  __attribute__((vector_size(16)));
typedef unsigned tdm_u4 __attribute__((vector_size(16)));
typedef int      tdm_i8 __attribute__((vector_size(32)));
typedef int      tdm_i4 __attribute__((vector_size(16)));

#if HAVE_TDM
# define A_INFLIGHT 4          // only A staged via async (4 b128 per thread)
#else
# define A_INFLIGHT 8          // A + B staged via async
#endif

// Round-to-nearest-even fp32 -> bf16, pack two into one uint (lo = a, hi = b)
__device__ __forceinline__ unsigned pack_bf16x2(float a, float b) {
    unsigned ua = __float_as_uint(a);
    ua = ua + 0x7fffu + ((ua >> 16) & 1u);
    unsigned ub = __float_as_uint(b);
    ub = ub + 0x7fffu + ((ub >> 16) & 1u);
    return ((ua >> 16) & 0xffffu) | (ub & 0xffff0000u);
}

// ---------------------------------------------------------------------------
// Kernel 0: bulk fp32 -> packed bf16 conversion (bandwidth bound, once/launch)
// ---------------------------------------------------------------------------
__global__ __launch_bounds__(256) void cvt_f32_to_bf16_kernel(
    const float4* __restrict__ in, uint2* __restrict__ outp, int n4)
{
    int i = blockIdx.x * 256 + threadIdx.x;
    if (i < n4) {
        float4 f = in[i];
        outp[i] = make_uint2(pack_bf16x2(f.x, f.y), pack_bf16x2(f.z, f.w));
    }
}

// ---------------------------------------------------------------------------
// Kernel 1: gate (mlp gate -> top2 -> softmax) ; one wave32 per token
// ---------------------------------------------------------------------------
__global__ __launch_bounds__(256) void moe_gate_kernel(
    const float* __restrict__ x, const float* __restrict__ w1,
    const float* __restrict__ w2, float* __restrict__ importance,
    unsigned* __restrict__ loadcnt, int* __restrict__ topidx,
    float* __restrict__ topscore)
{
    const int wave = threadIdx.x >> 5;
    const int lane = threadIdx.x & 31;
    const int t = blockIdx.x * 8 + wave;
    if (t >= TT) return;

    const float* xr = x + (size_t)t * DD;
    float acc[EE];
#pragma unroll
    for (int e = 0; e < EE; ++e) acc[e] = 0.f;

    for (int k = lane; k < DD; k += 32) {
        float xv = xr[k];
#pragma unroll
        for (int e = 0; e < EE; ++e) acc[e] += xv * w1[e * DD + k];
    }
#pragma unroll
    for (int e = 0; e < EE; ++e) {
#pragma unroll
        for (int off = 16; off > 0; off >>= 1)
            acc[e] += __shfl_xor(acc[e], off, 32);
    }

    if (lane == 0) {
        float th[EE], lg[EE];
#pragma unroll
        for (int e = 0; e < EE; ++e) th[e] = tanhf(acc[e]);
#pragma unroll
        for (int j = 0; j < EE; ++j) {
            float s = 0.f;
#pragma unroll
            for (int e = 0; e < EE; ++e) s += th[e] * w2[j * EE + e];
            lg[j] = s;
        }
        int j0 = 0;
#pragma unroll
        for (int j = 1; j < EE; ++j) if (lg[j] > lg[j0]) j0 = j;
        int j1 = (j0 == 0) ? 1 : 0;
#pragma unroll
        for (int j = 0; j < EE; ++j)
            if (j != j0 && lg[j] > lg[j1]) j1 = j;

        float e1 = __expf(lg[j1] - lg[j0]);
        float inv = 1.f / (1.f + e1);
        float s0 = inv, s1 = e1 * inv;

        topidx[2 * t + 0] = j0;  topidx[2 * t + 1] = j1;
        topscore[2 * t + 0] = s0; topscore[2 * t + 1] = s1;
        atomicAdd(&importance[j0], s0);
        atomicAdd(&importance[j1], s1);
        atomicAdd(&loadcnt[j0], 1u);
        atomicAdd(&loadcnt[j1], 1u);
    }
}

// ---------------------------------------------------------------------------
// Kernel 2: gate_loss (cv_squared, ddof=1) + bucket offsets
// ---------------------------------------------------------------------------
__global__ void moe_finalize_gate_kernel(
    const float* __restrict__ importance, const unsigned* __restrict__ loadcnt,
    unsigned* __restrict__ cursors, unsigned* __restrict__ offsets,
    float* __restrict__ out)
{
    if (threadIdx.x == 0 && blockIdx.x == 0) {
        float mi = 0.f, ml = 0.f;
#pragma unroll
        for (int e = 0; e < EE; ++e) { mi += importance[e]; ml += (float)loadcnt[e]; }
        mi *= (1.f / EE); ml *= (1.f / EE);
        float vi = 0.f, vl = 0.f;
#pragma unroll
        for (int e = 0; e < EE; ++e) {
            float di = importance[e] - mi; vi += di * di;
            float dl = (float)loadcnt[e] - ml; vl += dl * dl;
        }
        vi *= (1.f / (EE - 1)); vl *= (1.f / (EE - 1));
        out[(size_t)TT * OO] = BAL_W * (vi / (mi * mi + CV_EPS) + vl / (ml * ml + CV_EPS));

        unsigned off = 0;
#pragma unroll
        for (int e = 0; e < EE; ++e) {
            offsets[e] = off;
            cursors[e] = off;
            off += loadcnt[e];
        }
    }
}

// ---------------------------------------------------------------------------
// Kernel 3: scatter assignments into expert buckets
// ---------------------------------------------------------------------------
__global__ __launch_bounds__(256) void moe_scatter_kernel(
    const int* __restrict__ topidx, const float* __restrict__ topscore,
    unsigned* __restrict__ cursors, int* __restrict__ btok,
    float* __restrict__ bscore)
{
    int t = blockIdx.x * 256 + threadIdx.x;
    if (t >= TT) return;
#pragma unroll
    for (int k = 0; k < TOPK; ++k) {
        int e = topidx[2 * t + k];
        unsigned pos = atomicAdd(&cursors[e], 1u);
        btok[pos] = t;
        bscore[pos] = topscore[2 * t + k];
    }
}

// ---------------------------------------------------------------------------
// Kernel 4 (fast path): grouped expert GEMM on pre-staged bf16 via WMMA
// block = 256 threads = 8 waves (2 row-groups x 4 col-groups)
// block tile = 128 rows x 128 cols ; K-step 64 ; double-buffered LDS
// A staged with global_load_async_to_lds_b128 (ASYNCcnt)
// B staged with TDM tensor_load_to_lds (TENSORcnt), pad -> LDSTRIDE layout
// grid = (O/128, T/128, E)
// ---------------------------------------------------------------------------
#define LDSTRIDE 36   // 32 data uints + 4 pad; 144B, 16B aligned, bank-rotated
#define BUFBYTES (128u * LDSTRIDE * 4u)   // 18432 B per operand buffer

__global__ __launch_bounds__(256) void moe_gemm_bf16_kernel(
    const unsigned* __restrict__ xb,    // [T, D/2] packed bf16
    const unsigned* __restrict__ wb,    // [E, O, D/2] packed bf16
    const float* __restrict__ eb,       // [E, O]
    const unsigned* __restrict__ loadcnt,
    const unsigned* __restrict__ offsets,
    const int* __restrict__ btok,
    const float* __restrict__ bscore,
    float* __restrict__ out)            // [T, O]
{
    const int e = blockIdx.z;
    const unsigned cnt = loadcnt[e];
    const unsigned row0 = blockIdx.y * 128u;
    if (row0 >= cnt) return;
    const unsigned base = offsets[e];
    const int colBase = blockIdx.x * 128;

    __shared__ unsigned aU[2][128][LDSTRIDE];
    __shared__ unsigned bU[2][128][LDSTRIDE];
    __shared__ int   stok[128];
    __shared__ float sscore[128];

    const int tid = threadIdx.x;
    if (tid < 128) {
        unsigned r = row0 + (unsigned)tid;
        if (r < cnt) { stok[tid] = btok[base + r]; sscore[tid] = bscore[base + r]; }
        else         { stok[tid] = 0;              sscore[tid] = 0.f; }
    }
    __syncthreads();

    const int wave = tid >> 5;
    const int lane = tid & 31;
    const int nlo  = lane & 15;
    const int rw   = wave >> 2;          // 0..1 : row group (64 rows each)
    const int cw   = wave & 3;           // 0..3 : col group (32 cols each)
    const int halfOff = (lane >> 4) * 4; // CDNA5 16-bit operand K-half select

    const unsigned* wbase = wb + ((size_t)e * OO + (size_t)colBase) * D2;

    // Per-thread staging slots: 4 uint4 per operand tile, fixed rows/chunks.
    // LDS destinations kept as buffer-0 offsets; buffer select is a plain add
    // (avoids indexed-pointer arrays -> v_movrels in the hot loop).
    const unsigned* ag[4];
    const unsigned* bg[4];
    unsigned aoff[4], boff[4];
#pragma unroll
    for (int i = 0; i < 4; ++i) {
        int l = tid + i * 256, row = l >> 3, ch = l & 7;
        ag[i] = xb + (size_t)stok[row] * D2 + ch * 4;
        bg[i] = wbase + (size_t)row * D2 + ch * 4;
        aoff[i] = (unsigned)(size_t)&aU[0][row][ch * 4];
        boff[i] = (unsigned)(size_t)&bU[0][row][ch * 4];
    }

#if HAVE_TDM
    // D# for B tile: 128 rows x 32 DWORDs, row stride D2 DWORDs, LDS pad
    // interval 32 DWORDs (code 4) / amount 4 DWORDs (code 3), data_size 4B.
    const unsigned long long gbB = (unsigned long long)(size_t)wbase;
    const unsigned ldsB0 = (unsigned)(size_t)&bU[0][0][0];
    const int g1w0 = (int)((2u << 16) | (1u << 20) | (4u << 22) | (3u << 25));
    const int g1w1 = (int)(((unsigned)D2 & 0xffffu) << 16);        // dim0 lo16
    const int g1w2 = (int)((((unsigned)D2 >> 16) & 0xffffu) | (128u << 16));
    const int g1w3 = (int)(32u << 16);                             // tile_dim0
    const int g1w4 = 128;                                          // tile_dim1
    const int g1w5 = (int)(unsigned)D2;                            // dim0 stride
#endif

    auto stage = [&](int buf, int k0) {
        const unsigned bsel = (unsigned)buf * BUFBYTES;
        // A tile: token-gathered rows
#if HAVE_ASYNC
#pragma unroll
        for (int i = 0; i < 4; ++i)
            __builtin_amdgcn_global_load_async_to_lds_b128(
                (AS1 v4i_g*)(size_t)(ag[i] + k0),
                (AS3 v4i_g*)(aoff[i] + bsel), 0, 0);
#else
#pragma unroll
        for (int i = 0; i < 4; ++i) {
            int l = tid + i * 256, row = l >> 3, ch = l & 7;
            *(uint4*)&aU[buf][row][ch * 4] = *(const uint4*)(ag[i] + k0);
        }
#endif
        // B tile: dense 2D weight tile
#if HAVE_TDM
        if (wave == 0) {
            unsigned long long ga = gbB + (unsigned long long)k0 * 4ull;
            tdm_u4 g0v = { 1u,                                  // count=1
                           ldsB0 + bsel,                        // lds_addr
                           (unsigned)ga,                        // gaddr lo
                           (unsigned)((ga >> 32) & 0x01ffffffull) | 0x80000000u };
            tdm_i8 g1v = { g1w0, g1w1, g1w2, g1w3, g1w4, g1w5, 0, 0 };
            tdm_i4 z4  = { 0, 0, 0, 0 };
            tdm_i8 z8  = { 0, 0, 0, 0, 0, 0, 0, 0 };
            __builtin_amdgcn_tensor_load_to_lds(g0v, g1v, z4, z4, z8, 0);
        }
#elif HAVE_ASYNC
#pragma unroll
        for (int i = 0; i < 4; ++i)
            __builtin_amdgcn_global_load_async_to_lds_b128(
                (AS1 v4i_g*)(size_t)(bg[i] + k0),
                (AS3 v4i_g*)(boff[i] + bsel), 0, 0);
#else
#pragma unroll
        for (int i = 0; i < 4; ++i) {
            int l = tid + i * 256, row = l >> 3, ch = l & 7;
            *(uint4*)&bU[buf][row][ch * 4] = *(const uint4*)(bg[i] + k0);
        }
#endif
    };

    v8f c[4][2] = {};

    auto compute = [&](int buf) {
#pragma unroll
        for (int kc = 0; kc < 2; ++kc) {
            const int co = kc * 16 + halfOff;
            union { uint4 q[2]; v16bf v; } Af[4], Bf[2];
#pragma unroll
            for (int r = 0; r < 4; ++r) {
                const int ar = rw * 64 + r * 16 + nlo;
                Af[r].q[0] = *(const uint4*)&aU[buf][ar][co];
                Af[r].q[1] = *(const uint4*)&aU[buf][ar][co + 8];
            }
#pragma unroll
            for (int cc = 0; cc < 2; ++cc) {
                const int bc = cw * 32 + cc * 16 + nlo;
                Bf[cc].q[0] = *(const uint4*)&bU[buf][bc][co];
                Bf[cc].q[1] = *(const uint4*)&bU[buf][bc][co + 8];
            }
#pragma unroll
            for (int r = 0; r < 4; ++r)
#pragma unroll
                for (int cc = 0; cc < 2; ++cc)
                    c[r][cc] = __builtin_amdgcn_wmma_f32_16x16x32_bf16(
                        false, Af[r].v, false, Bf[cc].v, (short)0, c[r][cc],
                        false, false);
        }
    };

    stage(0, 0);
    for (int s = 0; s < NSTEP; ++s) {
        if (s + 1 < NSTEP) {
            stage((s + 1) & 1, (s + 1) * 32);
#if HAVE_ASYNC
            __builtin_amdgcn_s_wait_asynccnt(A_INFLIGHT);
#endif
#if HAVE_TDM
            __builtin_amdgcn_s_wait_tensorcnt(1);
#endif
        } else {
#if HAVE_ASYNC
            __builtin_amdgcn_s_wait_asynccnt(0);
#endif
#if HAVE_TDM
            __builtin_amdgcn_s_wait_tensorcnt(0);
#endif
        }
        __syncthreads();
        compute(s & 1);
        __syncthreads();
    }

    // epilogue: y[token, col] += score * (acc + bias[e, col])
    const int mb = (lane >= 16) ? 8 : 0;
#pragma unroll
    for (int cc = 0; cc < 2; ++cc) {
        const int col = colBase + cw * 32 + cc * 16 + nlo;
        const float bias = eb[e * OO + col];
#pragma unroll
        for (int r = 0; r < 4; ++r) {
#pragma unroll
            for (int i = 0; i < 8; ++i) {
                const int m = rw * 64 + r * 16 + mb + i;
                const float s = sscore[m];
                atomicAdd(out + (size_t)stok[m] * OO + col, s * (c[r][cc][i] + bias));
            }
        }
    }
}

// ---------------------------------------------------------------------------
// Kernel 4 (fallback, small ws): convert-in-kernel GEMM, 16x128 tile
// ---------------------------------------------------------------------------
__global__ __launch_bounds__(256) void moe_gemm_f32_kernel(
    const float* __restrict__ x, const float* __restrict__ ew,
    const float* __restrict__ eb, const unsigned* __restrict__ loadcnt,
    const unsigned* __restrict__ offsets, const int* __restrict__ btok,
    const float* __restrict__ bscore, float* __restrict__ out)
{
    const int e = blockIdx.z;
    const unsigned cnt = loadcnt[e];
    const unsigned row0 = blockIdx.y * 16u;
    if (row0 >= cnt) return;
    const unsigned base = offsets[e];
    const int colBase = blockIdx.x * 128;

    __shared__ unsigned aU[16][17];
    __shared__ unsigned bU[128][17];
    __shared__ int   stok[16];
    __shared__ float sscore[16];

    const int tid = threadIdx.x;
    if (tid < 16) {
        unsigned r = row0 + (unsigned)tid;
        if (r < cnt) { stok[tid] = btok[base + r]; sscore[tid] = bscore[base + r]; }
        else         { stok[tid] = 0;              sscore[tid] = 0.f; }
    }
    __syncthreads();

    const int wave = tid >> 5, lane = tid & 31, nlo = lane & 15;
    const int halfOff = (lane >> 4) * 4;
    v8f c = {};
    const float* wbase = ew + ((size_t)e * OO + (size_t)colBase) * DD;

    for (int k0 = 0; k0 < DD; k0 += 32) {
        if (tid < 128) {
            int row = tid >> 3, ch = tid & 7;
            const float4 f = *(const float4*)(x + (size_t)stok[row] * DD + k0 + ch * 4);
            aU[row][ch * 2 + 0] = pack_bf16x2(f.x, f.y);
            aU[row][ch * 2 + 1] = pack_bf16x2(f.z, f.w);
        }
#pragma unroll
        for (int i = 0; i < 4; ++i) {
            int lin = tid + i * 256;
            int row = lin >> 3, ch = lin & 7;
            const float4 f = *(const float4*)(wbase + (size_t)row * DD + k0 + ch * 4);
            bU[row][ch * 2 + 0] = pack_bf16x2(f.x, f.y);
            bU[row][ch * 2 + 1] = pack_bf16x2(f.z, f.w);
        }
        __syncthreads();

        union { unsigned u[8]; v16bf v; } A, Bf;
#pragma unroll
        for (int j = 0; j < 4; ++j) {
            A.u[j] = aU[nlo][halfOff + j];  A.u[4 + j] = aU[nlo][halfOff + 8 + j];
        }
        const int bcol = wave * 16 + nlo;
#pragma unroll
        for (int j = 0; j < 4; ++j) {
            Bf.u[j] = bU[bcol][halfOff + j]; Bf.u[4 + j] = bU[bcol][halfOff + 8 + j];
        }
        c = __builtin_amdgcn_wmma_f32_16x16x32_bf16(false, A.v, false, Bf.v,
                                                    (short)0, c, false, false);
        __syncthreads();
    }

    const int col = colBase + wave * 16 + nlo;
    const float bias = eb[e * OO + col];
    const int mb = (lane >= 16) ? 8 : 0;
#pragma unroll
    for (int r = 0; r < 8; ++r) {
        int m = mb + r;
        atomicAdd(out + (size_t)stok[m] * OO + col, sscore[m] * (c[r] + bias));
    }
}

// ---------------------------------------------------------------------------
// Launch
// ---------------------------------------------------------------------------
extern "C" void kernel_launch(void* const* d_in, const int* in_sizes, int n_in,
                              void* d_out, int out_size, void* d_ws, size_t ws_size,
                              hipStream_t stream) {
    const float* x   = (const float*)d_in[0];
    const float* w1  = (const float*)d_in[1];
    const float* w2  = (const float*)d_in[2];
    const float* ew  = (const float*)d_in[3];
    const float* eb  = (const float*)d_in[4];
    float* out = (float*)d_out;

    unsigned* wsu = (unsigned*)d_ws;
    float*    importance = (float*)wsu;            // 8
    unsigned* loadcnt    = wsu + 8;                // 8
    unsigned* cursors    = wsu + 16;               // 8
    unsigned* offsets    = wsu + 24;               // 8
    int*      topidx     = (int*)(wsu + 32);       // T*2
    float*    topscore   = (float*)(wsu + 32 + NASSIGN);
    int*      btok       = (int*)(wsu + 32 + 2 * NASSIGN);
    float*    bscore     = (float*)(wsu + 32 + 3 * NASSIGN);
    const size_t bookU   = 32 + 4 * (size_t)NASSIGN;
    unsigned* xb = wsu + bookU;                    // [T, D/2]
    unsigned* wb = xb + (size_t)TT * D2;           // [E, O, D/2]
    const size_t needBytes = bookU * 4 + (size_t)TT * DD * 2
                           + (size_t)EE * OO * DD * 2;

    (void)hipMemsetAsync(d_out, 0, (size_t)out_size * sizeof(float), stream);
    (void)hipMemsetAsync(d_ws, 0, 32 * sizeof(unsigned), stream);

    moe_gate_kernel<<<TT / 8, 256, 0, stream>>>(x, w1, w2, importance, loadcnt,
                                                topidx, topscore);
    moe_finalize_gate_kernel<<<1, 32, 0, stream>>>(importance, loadcnt, cursors,
                                                   offsets, out);
    moe_scatter_kernel<<<TT / 256, 256, 0, stream>>>(topidx, topscore, cursors,
                                                     btok, bscore);

    if (ws_size >= needBytes) {
        const int n4x = TT * DD / 4;
        const int n4w = EE * OO * DD / 4;
        cvt_f32_to_bf16_kernel<<<n4x / 256, 256, 0, stream>>>(
            (const float4*)x, (uint2*)xb, n4x);
        cvt_f32_to_bf16_kernel<<<n4w / 256, 256, 0, stream>>>(
            (const float4*)ew, (uint2*)wb, n4w);
        dim3 grid(OO / 128, TT / 128, EE);
        moe_gemm_bf16_kernel<<<grid, 256, 0, stream>>>(
            xb, wb, eb, loadcnt, offsets, btok, bscore, out);
    } else {
        dim3 grid(OO / 128, TT / 16, EE);
        moe_gemm_f32_kernel<<<grid, 256, 0, stream>>>(
            x, ew, eb, loadcnt, offsets, btok, bscore, out);
    }
}
